// DownBlock_2817498546729
// MI455X (gfx1250) — compile-verified, matrix operand order checked by reading, not run on hardware
//
#include <hip/hip_runtime.h>
#include <hip/hip_bf16.h>

// ---------------- problem constants ----------------
#define BATCH   32
#define LSEQ    196
#define MROWS   (BATCH*LSEQ)   // 6272 rows = 98 * 64
#define DMODEL  512
#define DINNER  1024
#define DSTATE  16
#define DTRANK  32
#define KCONV   4
#define NLAYER  4
#define MT      4              // 16-row M tiles per wave (64 rows)

typedef __attribute__((ext_vector_type(16))) __bf16 v16bf;
typedef __attribute__((ext_vector_type(8)))  float  v8f;

__device__ __forceinline__ unsigned short f2bf(float f) {
  union { float f; unsigned int u; } x; x.f = f;
  unsigned int r = x.u + 0x7fffu + ((x.u >> 16) & 1u);  // RNE
  return (unsigned short)(r >> 16);
}
__device__ __forceinline__ float sigmoidf_(float x) { return 1.0f / (1.0f + __expf(-x)); }

// ---------------- elementwise: f32 -> bf16 convert ----------------
__global__ void cvt_kernel(const float* __restrict__ src, unsigned short* __restrict__ dst, int n) {
  int i = blockIdx.x * blockDim.x + threadIdx.x;
  if (i < n) dst[i] = f2bf(src[i]);
}

// ---------------- max-pool pairs along L, emit bf16 ----------------
__global__ void pool_kernel(const float* __restrict__ src, unsigned short* __restrict__ dst) {
  int i = blockIdx.x * blockDim.x + threadIdx.x;
  if (i >= MROWS * DMODEL) return;
  int c = i & (DMODEL - 1);
  int row = i >> 9;                 // /512
  int b = row / LSEQ, t = row - b * LSEQ;
  const float* p = src + (size_t)(b * 392 + 2 * t) * DMODEL + c;
  dst[i] = f2bf(fmaxf(p[0], p[DMODEL]));
}

// ---------------- WMMA bf16 GEMM: C[M,N] = act(A[M,K] * W[N,K]^T + bias + rowvec [+ C]) -------
// A/B fragment loader per CDNA5 16-bit 16x32 layout:
//   lane L: row = L&15 ; K dwords: lower half-wave K = k0+{0..7,16..23}, upper K = k0+{8..15,24..31}
__device__ __forceinline__ v16bf load_frag(const unsigned short* __restrict__ p,
                                           int ld, int row16, int k0, int lane) {
  const unsigned short* r = p + (size_t)(row16 + (lane & 15)) * ld + k0 + ((lane >> 4) << 3);
  union { v16bf v; unsigned int u[8]; } f;
#pragma unroll
  for (int i = 0; i < 4; ++i) f.u[i]     = *(const unsigned int*)(r + 2 * i);
#pragma unroll
  for (int i = 0; i < 4; ++i) f.u[4 + i] = *(const unsigned int*)(r + 16 + 2 * i);
  return f.v;
}

// One wave computes a 64x16 (MT M-tiles x one N-tile) block: the W fragment is
// reused MT times per K step and the MT accumulator chains are independent.
__global__ void gemm_bf16_kernel(const unsigned short* __restrict__ A, int lda,
                                 const unsigned short* __restrict__ W, int ldw,
                                 const float* __restrict__ bias,
                                 const float* __restrict__ rowvec, int rowdiv,
                                 float* __restrict__ Cf, unsigned short* __restrict__ Cb,
                                 int ldc, int N, int K, int accumulate, int act) {
  int lane = threadIdx.x & 31;
  int wave = threadIdx.x >> 5;
  int tn = blockIdx.x * 8 + wave;       // one 16-col N tile per wave
  if (tn * 16 >= N) return;             // wave-uniform exit (EXEC all-ones at WMMA)
  int m0 = blockIdx.y * (MT * 16);

  v8f acc[MT] = {};
  for (int k0 = 0; k0 < K; k0 += 32) {
    v16bf b = load_frag(W, ldw, tn * 16, k0, lane);
#pragma unroll
    for (int i = 0; i < MT; ++i) {
      v16bf a = load_frag(A, lda, m0 + i * 16, k0, lane);
      acc[i] = __builtin_amdgcn_wmma_f32_16x16x32_bf16(false, a, false, b, (short)0,
                                                       acc[i], false, false);
    }
  }

  // C/D layout: VGPR r -> M = r + 8*(lane>=16), N = lane&15
  int n = tn * 16 + (lane & 15);
  float bn = bias ? bias[n] : 0.0f;
#pragma unroll
  for (int i = 0; i < MT; ++i) {
    int mbase = m0 + i * 16 + ((lane >> 4) << 3);
#pragma unroll
    for (int r = 0; r < 8; ++r) {
      int m = mbase + r;
      size_t off = (size_t)m * ldc + n;
      float v = acc[i][r] + bn;
      if (rowvec)     v += rowvec[(size_t)(m / rowdiv) * N + n];
      if (accumulate) v += Cf[off];
      if (act == 1)   v = (v > 20.0f) ? v : log1pf(__expf(v));   // softplus
      if (Cf) Cf[off] = v;
      if (Cb) Cb[off] = f2bf(v);
    }
  }
}

// ---------------- silu(rmsnorm(x)) over rows of 512, bf16 out ----------------
__global__ void rmsnorm_silu_kernel(const float* __restrict__ X, const float* __restrict__ w,
                                    unsigned short* __restrict__ out) {
  __shared__ float red[256];
  int row = blockIdx.x;
  const float* x = X + (size_t)row * DMODEL;
  float a = x[threadIdx.x], c = x[threadIdx.x + 256];
  red[threadIdx.x] = a * a + c * c;
  __syncthreads();
  for (int s = 128; s > 0; s >>= 1) {
    if (threadIdx.x < s) red[threadIdx.x] += red[threadIdx.x + s];
    __syncthreads();
  }
  float inv = rsqrtf(red[0] / DMODEL + 1e-5f);
  float v0 = a * inv * w[threadIdx.x];
  float v1 = c * inv * w[threadIdx.x + 256];
  out[(size_t)row * DMODEL + threadIdx.x]       = f2bf(v0 * sigmoidf_(v0));
  out[(size_t)row * DMODEL + threadIdx.x + 256] = f2bf(v1 * sigmoidf_(v1));
}

// ---------------- layernorm over rows of 512; bf16 and/or f32 out ----------------
__global__ void layernorm_kernel(const float* __restrict__ X, const float* __restrict__ w,
                                 const float* __restrict__ b,
                                 unsigned short* __restrict__ out_bf, float* __restrict__ out_f) {
  __shared__ float rs[256], rq[256];
  int row = blockIdx.x;
  const float* x = X + (size_t)row * DMODEL;
  float a = x[threadIdx.x], c = x[threadIdx.x + 256];
  rs[threadIdx.x] = a + c;
  rq[threadIdx.x] = a * a + c * c;
  __syncthreads();
  for (int s = 128; s > 0; s >>= 1) {
    if (threadIdx.x < s) { rs[threadIdx.x] += rs[threadIdx.x + s]; rq[threadIdx.x] += rq[threadIdx.x + s]; }
    __syncthreads();
  }
  float mu  = rs[0] / DMODEL;
  float var = rq[0] / DMODEL - mu * mu;
  float inv = rsqrtf(var + 1e-5f);
  float v0 = (a - mu) * inv * w[threadIdx.x]       + b[threadIdx.x];
  float v1 = (c - mu) * inv * w[threadIdx.x + 256] + b[threadIdx.x + 256];
  if (out_bf) {
    out_bf[(size_t)row * DMODEL + threadIdx.x]       = f2bf(v0);
    out_bf[(size_t)row * DMODEL + threadIdx.x + 256] = f2bf(v1);
  }
  if (out_f) {
    out_f[(size_t)row * DMODEL + threadIdx.x]       = v0;
    out_f[(size_t)row * DMODEL + threadIdx.x + 256] = v1;
  }
}

// ---------------- causal depthwise conv K=4 + bias + silu ----------------
__global__ void conv_silu_kernel(const float* __restrict__ XZ, const float* __restrict__ cw,
                                 const float* __restrict__ cb,
                                 float* __restrict__ XI, unsigned short* __restrict__ XIb) {
  int i = blockIdx.x * blockDim.x + threadIdx.x;
  if (i >= MROWS * DINNER) return;
  int d = i & (DINNER - 1);
  int row = i >> 10;
  int t = row % LSEQ;
  const float* wk = cw + d * KCONV;
  float acc = 0.0f;
#pragma unroll
  for (int k = 0; k < KCONV; ++k) {
    int tt = t - (KCONV - 1) + k;
    if (tt >= 0) acc += XZ[(size_t)(row - (KCONV - 1) + k) * (2 * DINNER) + d] * wk[k];
  }
  acc += cb[d];
  float s = acc * sigmoidf_(acc);
  XI[i] = s;
  XIb[i] = f2bf(s);
}

// ---------------- selective scan: one thread per channel, 16 fp32 states in regs ----------------
__global__ void scan_kernel(const float* __restrict__ DT, const float* __restrict__ XI,
                            const float* __restrict__ DBC, const float* __restrict__ A_log,
                            float* __restrict__ Y) {
  int b = blockIdx.x;
  int d = blockIdx.y * 256 + threadIdx.x;
  __shared__ float sBC[32];                 // B[16] then C[16] for current t
  float A[DSTATE], h[DSTATE];
#pragma unroll
  for (int n = 0; n < DSTATE; ++n) { A[n] = -__expf(A_log[(size_t)d * DSTATE + n]); h[n] = 0.0f; }
  for (int t = 0; t < LSEQ; ++t) {
    int row = b * LSEQ + t;
    __syncthreads();
    if (threadIdx.x < 32) sBC[threadIdx.x] = DBC[(size_t)row * 64 + 32 + threadIdx.x];
    __syncthreads();
    float dt = DT[(size_t)row * DINNER + d];
    float dx = dt * XI[(size_t)row * DINNER + d];
    float y = 0.0f;
#pragma unroll
    for (int n = 0; n < DSTATE; ++n) {
      h[n] = __expf(dt * A[n]) * h[n] + dx * sBC[n];
      y += h[n] * sBC[16 + n];
    }
    Y[(size_t)row * DINNER + d] = y;
  }
}

// ---------------- y = (scan + D*xi) * silu(z)  -> bf16 ----------------
__global__ void gate_kernel(const float* __restrict__ Y, const float* __restrict__ XI,
                            const float* __restrict__ XZ, const float* __restrict__ Dp,
                            unsigned short* __restrict__ Yb) {
  int i = blockIdx.x * blockDim.x + threadIdx.x;
  if (i >= MROWS * DINNER) return;
  int d = i & (DINNER - 1);
  int row = i >> 10;
  float y = Y[i] + Dp[d] * XI[i];
  float z = XZ[(size_t)row * (2 * DINNER) + DINNER + d];
  Yb[i] = f2bf(y * z * sigmoidf_(z));
}

// ================= host orchestration =================
extern "C" void kernel_launch(void* const* d_in, const int* in_sizes, int n_in,
                              void* d_out, int out_size, void* d_ws, size_t ws_size,
                              hipStream_t stream) {
  (void)in_sizes; (void)n_in; (void)out_size; (void)ws_size;
  const float* motion   = (const float*)d_in[0];
  const float* embed    = (const float*)d_in[1];
  const float* mlp_w1   = (const float*)d_in[2];
  const float* mlp_b1   = (const float*)d_in[3];
  const float* mlp_rmsw = (const float*)d_in[4];
  const float* mlp_w2   = (const float*)d_in[5];
  const float* mlp_b2   = (const float*)d_in[6];
  const float* ln_w     = (const float*)d_in[7];
  const float* ln_b     = (const float*)d_in[8];
  const float* in_w     = (const float*)d_in[9];
  const float* conv_w   = (const float*)d_in[10];
  const float* conv_b   = (const float*)d_in[11];
  const float* xproj_w  = (const float*)d_in[12];
  const float* dt_w     = (const float*)d_in[13];
  const float* dt_b     = (const float*)d_in[14];
  const float* A_log    = (const float*)d_in[15];
  const float* D_param  = (const float*)d_in[16];
  const float* out_w    = (const float*)d_in[17];
  const float* lnf_w    = (const float*)d_in[18];
  const float* lnf_b    = (const float*)d_in[19];

  char* ws = (char*)d_ws;
  size_t off = 0;
  auto alloc = [&](size_t bytes) -> char* {
    char* p = ws + off; off = (off + bytes + 255) & ~(size_t)255; return p;
  };
  // bf16 weights
  unsigned short* wb_mlp1  = (unsigned short*)alloc((size_t)DMODEL * DMODEL * 2);
  unsigned short* wb_mlp2  = (unsigned short*)alloc((size_t)DMODEL * DMODEL * 2);
  unsigned short* wb_in    = (unsigned short*)alloc((size_t)NLAYER * 2 * DINNER * DMODEL * 2);
  unsigned short* wb_xproj = (unsigned short*)alloc((size_t)NLAYER * 64 * DINNER * 2);
  unsigned short* wb_dt    = (unsigned short*)alloc((size_t)NLAYER * DINNER * DTRANK * 2);
  unsigned short* wb_out   = (unsigned short*)alloc((size_t)NLAYER * DMODEL * DINNER * 2);
  // activations (with aliasing of pre-loop temporaries onto loop buffers)
  float*          Xbuf  = (float*)alloc((size_t)MROWS * DMODEL * 4);           // residual stream
  unsigned short* Xn    = (unsigned short*)alloc((size_t)MROWS * DMODEL * 2);  // layernorm bf16
  float*          XZ    = (float*)alloc((size_t)MROWS * 2 * DINNER * 4);       // in_proj out
  float*          XI    = (float*)alloc((size_t)MROWS * DINNER * 4);           // silu(conv) f32
  unsigned short* XIb   = (unsigned short*)alloc((size_t)MROWS * DINNER * 2);  // bf16 mirror
  float*          DBC   = (float*)alloc((size_t)MROWS * 64 * 4);
  unsigned short* DBCb  = (unsigned short*)alloc((size_t)MROWS * 64 * 2);
  float*          DTb   = (float*)alloc((size_t)MROWS * DINNER * 4);
  float*          Ybuf  = (float*)alloc((size_t)MROWS * DINNER * 4);
  unsigned short* Yb    = (unsigned short*)alloc((size_t)MROWS * DINNER * 2);
  // aliases (used only before the layer loop)
  unsigned short* Xp = Yb;                  // pooled bf16 input (M*512 <= M*1024)
  float*          T1 = DTb;                 // mlp hidden f32   (M*512 <= M*1024)
  unsigned short* Hb = XIb;                 // silu(rmsnorm) bf16

  auto cvt = [&](const float* s, unsigned short* d, int n) {
    cvt_kernel<<<(n + 255) / 256, 256, 0, stream>>>(s, d, n);
  };
  auto gemm = [&](const unsigned short* A, int lda, const unsigned short* W, int ldw,
                  const float* bias, const float* rowvec, int rowdiv,
                  float* Cf, unsigned short* Cb, int ldc, int N, int K, int accum, int act) {
    dim3 grid((N + 127) / 128, MROWS / (MT * 16));
    gemm_bf16_kernel<<<grid, 256, 0, stream>>>(A, lda, W, ldw, bias, rowvec, rowdiv,
                                               Cf, Cb, ldc, N, K, accum, act);
  };

  // weights -> bf16 (once per launch; deterministic, capture-safe)
  cvt(mlp_w1, wb_mlp1, DMODEL * DMODEL);
  cvt(mlp_w2, wb_mlp2, DMODEL * DMODEL);
  cvt(in_w,    wb_in,    NLAYER * 2 * DINNER * DMODEL);
  cvt(xproj_w, wb_xproj, NLAYER * 64 * DINNER);
  cvt(dt_w,    wb_dt,    NLAYER * DINNER * DTRANK);
  cvt(out_w,   wb_out,   NLAYER * DMODEL * DINNER);

  // stem: pool -> mlp1 -> silu(rmsnorm) -> mlp2 (+b2 +embed) = residual X
  pool_kernel<<<(MROWS * DMODEL + 255) / 256, 256, 0, stream>>>(motion, Xp);
  gemm(Xp, DMODEL, wb_mlp1, DMODEL, mlp_b1, nullptr, 1, T1, nullptr, DMODEL, DMODEL, DMODEL, 0, 0);
  rmsnorm_silu_kernel<<<MROWS, 256, 0, stream>>>(T1, mlp_rmsw, Hb);
  gemm(Hb, DMODEL, wb_mlp2, DMODEL, mlp_b2, embed, LSEQ, Xbuf, nullptr, DMODEL, DMODEL, DMODEL, 0, 0);

  for (int l = 0; l < NLAYER; ++l) {
    layernorm_kernel<<<MROWS, 256, 0, stream>>>(Xbuf, ln_w + l * DMODEL, ln_b + l * DMODEL, Xn, nullptr);
    gemm(Xn, DMODEL, wb_in + (size_t)l * 2 * DINNER * DMODEL, DMODEL,
         nullptr, nullptr, 1, XZ, nullptr, 2 * DINNER, 2 * DINNER, DMODEL, 0, 0);
    conv_silu_kernel<<<(MROWS * DINNER + 255) / 256, 256, 0, stream>>>(
        XZ, conv_w + (size_t)l * DINNER * KCONV, conv_b + (size_t)l * DINNER, XI, XIb);
    // xproj: f32 out (B,C for scan) + fused bf16 mirror (dt GEMM input)
    gemm(XIb, DINNER, wb_xproj + (size_t)l * 64 * DINNER, DINNER,
         nullptr, nullptr, 1, DBC, DBCb, 64, 64, DINNER, 0, 0);
    // dt proj with fused softplus epilogue
    gemm(DBCb, 64, wb_dt + (size_t)l * DINNER * DTRANK, DTRANK,
         dt_b + (size_t)l * DINNER, nullptr, 1, DTb, nullptr, DINNER, DINNER, DTRANK, 0, 1);
    scan_kernel<<<dim3(BATCH, DINNER / 256), 256, 0, stream>>>(
        DTb, XI, DBC, A_log + (size_t)l * DINNER * DSTATE, Ybuf);
    gate_kernel<<<(MROWS * DINNER + 255) / 256, 256, 0, stream>>>(
        Ybuf, XI, XZ, D_param + (size_t)l * DINNER, Yb);
    gemm(Yb, DINNER, wb_out + (size_t)l * DMODEL * DINNER, DINNER,
         nullptr, nullptr, 1, Xbuf, nullptr, DMODEL, DMODEL, DINNER, 1, 0);  // X += y @ out_w^T
  }

  layernorm_kernel<<<MROWS, 256, 0, stream>>>(Xbuf, lnf_w, lnf_b, nullptr, (float*)d_out);
}
